// ConformerEnsembleEmbeddingCombiner_82368882802831
// MI455X (gfx1250) — matrix operand order repaired and checked by reading, not compiled, not run on hardware
//
#include <hip/hip_runtime.h>
#include <hip/hip_bf16.h>
#include <stdint.h>

typedef __attribute__((ext_vector_type(16))) _Float16 v16h;
typedef __attribute__((ext_vector_type(8)))  float    v8f;

#define G_GROUPS 8192
#define GPB      4            // groups per block
#define NBLOCKS  (G_GROUPS / GPB)
#define BLOCK    256          // 8 wave32s
#define S_ROWS   32
#define MROWS    (GPB * S_ROWS)   // 128 rows per block
#define SDIM     128
#define VDIM     384
#define INDIM    512

// output float offsets: mp_s | mp_v | ds_s | ds_v | sa_s | sa_v
#define OUT_MP_S 0UL
#define OUT_MP_V 1048576UL
#define OUT_DS_S 4194304UL
#define OUT_DS_V 5242880UL
#define OUT_SA_S 8388608UL
#define OUT_SA_V 9437184UL

struct CombParams {
  const float* x;
  // f16 weights (pre-converted into workspace) for WMMA GEMMs
  const _Float16 *w_ds_phi_s0, *w_ds_phi_s1, *w_sa_phi_s0, *w_sa_phi_s1, *w_attn_s;
  const _Float16 *w_ds_phi_v0, *w_ds_phi_v1, *w_sa_phi_v0, *w_sa_phi_v1, *w_attn_v;
  // f32 biases for GEMM epilogues
  const float *b_ds_phi_s0, *b_ds_phi_s1, *b_sa_phi_s0, *b_sa_phi_s1, *b_attn_s;
  const float *b_ds_phi_v0, *b_ds_phi_v1, *b_sa_phi_v0, *b_sa_phi_v1, *b_attn_v;
  // rho MLPs (tiny per-group vectors, fp32 VALU straight from global params)
  const float *w_ds_rho_s0, *bb_ds_rho_s0, *w_ds_rho_s1, *bb_ds_rho_s1;
  const float *w_ds_rho_v0, *bb_ds_rho_v0, *w_ds_rho_v1, *bb_ds_rho_v1;
  const float *w_sa_rho_s0, *bb_sa_rho_s0, *w_sa_rho_s1, *bb_sa_rho_s1;
  const float *w_sa_rho_v0, *bb_sa_rho_v0, *w_sa_rho_v1, *bb_sa_rho_v1;
  float* out;
};

union FragU { v16h v; uint32_t u[8]; };

__device__ __forceinline__ float silu_f(float x) {
  // x * sigmoid(x) with single v_rcp_f32 instead of IEEE division chain
  return x * __builtin_amdgcn_rcpf(1.0f + __expf(-x));
}

// A fragment: 16x32 f16, CDNA5 wave32 layout (lane m = l&15; lanes>=16 shift K by 8 per half)
__device__ __forceinline__ v16h load_a_frag(const _Float16* A, int lda, int mbase, int kbase, int lane) {
  FragU f;
  const int m   = mbase + (lane & 15);
  const int kh8 = (lane >> 4) << 3;
  const _Float16* row = A + m * lda + kbase;
#pragma unroll
  for (int j = 0; j < 4; ++j)
    f.u[j] = *(const uint32_t*)(row + kh8 + 2 * j);
#pragma unroll
  for (int j = 0; j < 4; ++j)
    f.u[4 + j] = *(const uint32_t*)(row + 16 + kh8 + 2 * j);
  return f.v;
}

// B fragment: 32x16 f16 (K x N), B[k][n] = W[n][k], W row-major (dout x din)
// lane n = l&15; lanes>=16 shift K by 8 per half (same K order as A fragment)
__device__ __forceinline__ v16h load_b_frag(const _Float16* W, int ldw, int nbase, int kbase, int lane) {
  FragU f;
  const int n    = nbase + (lane & 15);
  const int kh8  = (lane >> 4) << 3;
  const _Float16* row = W + n * ldw + kbase;
#pragma unroll
  for (int j = 0; j < 4; ++j)
    f.u[j] = *(const uint32_t*)(row + kh8 + 2 * j);
#pragma unroll
  for (int j = 0; j < 4; ++j)
    f.u[4 + j] = *(const uint32_t*)(row + 16 + kh8 + 2 * j);
  return f.v;
}

// OUT(128xN) = act(A(128xK) @ W^T + b)   A,OUT in LDS f16; W f16 (L2-resident); b f32
// 8 waves: wave = M-tile (16 rows each); every wave walks all N-tiles
__device__ __forceinline__ void gemm_tile(_Float16* OUT, const _Float16* A,
                                          const _Float16* W, const float* bias,
                                          int N, int K, bool dosilu,
                                          int wave, int lane) {
  const int mbase = wave << 4;
  __builtin_prefetch(W, 0, 1);  // global_prefetch_b8: pull weight stream toward WGP
  for (int nt = 0; nt < (N >> 4); ++nt) {
    const int nbase = nt << 4;
    v8f acc = {};
    for (int kb = 0; kb < K; kb += 32) {
      v16h a = load_a_frag(A, K, mbase, kb, lane);
      v16h b = load_b_frag(W, K, nbase, kb, lane);
      acc = __builtin_amdgcn_wmma_f32_16x16x32_f16(false, a, false, b,
                                                   (short)0, acc, false, false);
    }
    const int n  = nbase + (lane & 15);
    const int mr = mbase + ((lane >> 4) << 3);
    const float bv = bias[n];
#pragma unroll
    for (int r = 0; r < 8; ++r) {
      float x = acc[r] + bv;
      if (dosilu) x = silu_f(x);
      OUT[(mr + r) * N + n] = (_Float16)x;
    }
  }
}

__device__ __forceinline__ float dotN(const float* __restrict__ w,
                                      const float* __restrict__ v, int n) {
  float a = 0.0f;
  for (int k = 0; k < n; ++k) a = fmaf(w[k], v[k], a);
  return a;
}

__global__ __launch_bounds__(BLOCK)
void conformer_combiner_kernel(CombParams P) {
  // 256 KB of f16 GEMM arenas + ~38 KB fp32 scratch  (< 320 KB/WGP)
  __shared__ _Float16 S0[MROWS * SDIM];   // 32 KB
  __shared__ _Float16 S1[MROWS * SDIM];   // 32 KB
  __shared__ _Float16 V0[MROWS * VDIM];   // 96 KB
  __shared__ _Float16 V1[MROWS * VDIM];   // 96 KB
  __shared__ float Ssc[GPB * S_ROWS * S_ROWS];
  __shared__ float meanS[GPB * SDIM];
  __shared__ float meanV[GPB * VDIM];
  __shared__ float rtmp[GPB * VDIM];
  __shared__ float wcol[GPB * S_ROWS];
  __shared__ float aggS[GPB * SDIM];
  __shared__ float aggV[GPB * VDIM];

  const int tid  = threadIdx.x;
  const int wave = tid >> 5;
  const int lane = tid & 31;
  const int g0   = blockIdx.x * GPB;
  const size_t row0 = (size_t)blockIdx.x * MROWS;

  // ---- 1) load 128x512 tile -> f16 LDS (S0 = scalar, V0 = vector), mean-pool out ----
  for (int c = tid; c < INDIM; c += BLOCK) {
    for (int gi = 0; gi < GPB; ++gi) {
      float s = 0.0f;
      for (int r = 0; r < S_ROWS; ++r) {
        const int rr = gi * S_ROWS + r;
        float v = P.x[(row0 + rr) * INDIM + c];
        s += v;
        if (c < SDIM) S0[rr * SDIM + c] = (_Float16)v;
        else          V0[rr * VDIM + (c - SDIM)] = (_Float16)v;
      }
      const float m = s * (1.0f / 32.0f);
      if (c < SDIM) P.out[OUT_MP_S + (size_t)(g0 + gi) * SDIM + c] = m;
      else          P.out[OUT_MP_V + (size_t)(g0 + gi) * VDIM + (c - SDIM)] = m;
    }
  }
  __syncthreads();

  // ---- 2) deep-sets scalar chain: S0(X) -> S1 -> S0(ds2) ----
  gemm_tile(S1, S0, P.w_ds_phi_s0, P.b_ds_phi_s0, SDIM, SDIM, true, wave, lane);
  __syncthreads();
  gemm_tile(S0, S1, P.w_ds_phi_s1, P.b_ds_phi_s1, SDIM, SDIM, true, wave, lane);
  __syncthreads();
  for (int idx = tid; idx < GPB * SDIM; idx += BLOCK) {   // per-group mean
    const int gi = idx >> 7, c = idx & (SDIM - 1);
    float s = 0.0f;
    for (int r = 0; r < S_ROWS; ++r) s += (float)S0[(gi * S_ROWS + r) * SDIM + c];
    meanS[idx] = s * (1.0f / 32.0f);
  }
  __syncthreads();
  for (int idx = tid; idx < GPB * SDIM; idx += BLOCK) {
    const int gi = idx >> 7, n = idx & (SDIM - 1);
    rtmp[idx] = silu_f(dotN(P.w_ds_rho_s0 + n * SDIM, meanS + gi * SDIM, SDIM) + P.bb_ds_rho_s0[n]);
  }
  __syncthreads();
  for (int idx = tid; idx < GPB * SDIM; idx += BLOCK) {
    const int gi = idx >> 7, n = idx & (SDIM - 1);
    P.out[OUT_DS_S + (size_t)(g0 + gi) * SDIM + n] =
        silu_f(dotN(P.w_ds_rho_s1 + n * SDIM, rtmp + gi * SDIM, SDIM) + P.bb_ds_rho_s1[n]);
  }
  __syncthreads();

  // ---- 3) deep-sets vector chain: V0(X) -> V1 -> V0(ds2v) ----
  gemm_tile(V1, V0, P.w_ds_phi_v0, P.b_ds_phi_v0, VDIM, VDIM, true, wave, lane);
  __syncthreads();
  gemm_tile(V0, V1, P.w_ds_phi_v1, P.b_ds_phi_v1, VDIM, VDIM, true, wave, lane);
  __syncthreads();
  for (int idx = tid; idx < GPB * VDIM; idx += BLOCK) {
    const int gi = idx / VDIM, c = idx % VDIM;
    float s = 0.0f;
    for (int r = 0; r < S_ROWS; ++r) s += (float)V0[(gi * S_ROWS + r) * VDIM + c];
    meanV[idx] = s * (1.0f / 32.0f);
  }
  __syncthreads();
  for (int idx = tid; idx < GPB * VDIM; idx += BLOCK) {
    const int gi = idx / VDIM, n = idx % VDIM;
    rtmp[idx] = silu_f(dotN(P.w_ds_rho_v0 + n * VDIM, meanV + gi * VDIM, VDIM) + P.bb_ds_rho_v0[n]);
  }
  __syncthreads();
  for (int idx = tid; idx < GPB * VDIM; idx += BLOCK) {
    const int gi = idx / VDIM, n = idx % VDIM;
    P.out[OUT_DS_V + (size_t)(g0 + gi) * VDIM + n] =
        silu_f(dotN(P.w_ds_rho_v1 + n * VDIM, rtmp + gi * VDIM, VDIM) + P.bb_ds_rho_v1[n]);
  }
  __syncthreads();

  // ---- 4) reload X -> S1 / V1 for the self-attention branch ----
  for (int c = tid; c < INDIM; c += BLOCK) {
    for (int rr = 0; rr < MROWS; ++rr) {
      float v = P.x[(row0 + rr) * INDIM + c];
      if (c < SDIM) S1[rr * SDIM + c] = (_Float16)v;
      else          V1[rr * VDIM + (c - SDIM)] = (_Float16)v;
    }
  }
  __syncthreads();

  // ---- 5) sa scalar: S1(X)->S0->S1(xs)->S0(qs);  sa vector: V1(X)->V0->V1(xv)->V0(qv) ----
  gemm_tile(S0, S1, P.w_sa_phi_s0, P.b_sa_phi_s0, SDIM, SDIM, true, wave, lane);
  __syncthreads();
  gemm_tile(S1, S0, P.w_sa_phi_s1, P.b_sa_phi_s1, SDIM, SDIM, true, wave, lane);  // xs
  __syncthreads();
  gemm_tile(S0, S1, P.w_attn_s, P.b_attn_s, SDIM, SDIM, false, wave, lane);       // qs
  __syncthreads();
  gemm_tile(V0, V1, P.w_sa_phi_v0, P.b_sa_phi_v0, VDIM, VDIM, true, wave, lane);
  __syncthreads();
  gemm_tile(V1, V0, P.w_sa_phi_v1, P.b_sa_phi_v1, VDIM, VDIM, true, wave, lane);  // xv
  __syncthreads();
  gemm_tile(V0, V1, P.w_attn_v, P.b_attn_v, VDIM, VDIM, false, wave, lane);       // qv
  __syncthreads();

  // ---- 6) scores = qs qs^T + qv qv^T  (4 groups x 32x32; 16 tiles over 8 waves) ----
#pragma unroll
  for (int i = 0; i < 2; ++i) {
    const int t  = wave * 2 + i;
    const int gi = t >> 2, mt = (t >> 1) & 1, nt = t & 1;
    const int arow = gi * S_ROWS + (mt << 4);
    const int brow = gi * S_ROWS + (nt << 4);
    v8f acc = {};
    for (int kb = 0; kb < SDIM; kb += 32) {
      v16h a = load_a_frag(S0, SDIM, arow, kb, lane);
      v16h b = load_b_frag(S0, SDIM, brow, kb, lane);  // B[k][n] = qs[n][k]
      acc = __builtin_amdgcn_wmma_f32_16x16x32_f16(false, a, false, b, (short)0, acc, false, false);
    }
    for (int kb = 0; kb < VDIM; kb += 32) {
      v16h a = load_a_frag(V0, VDIM, arow, kb, lane);
      v16h b = load_b_frag(V0, VDIM, brow, kb, lane);
      acc = __builtin_amdgcn_wmma_f32_16x16x32_f16(false, a, false, b, (short)0, acc, false, false);
    }
    const int n  = (nt << 4) + (lane & 15);
    const int mr = (mt << 4) + ((lane >> 4) << 3);
#pragma unroll
    for (int r = 0; r < 8; ++r)
      Ssc[gi * (S_ROWS * S_ROWS) + (mr + r) * S_ROWS + n] = acc[r];
  }
  __syncthreads();

  // ---- 7) row softmax, then column weights w[j] = sum_i attn[i][j] ----
  if (tid < GPB * S_ROWS) {
    const int gi = tid >> 5, row = tid & 31;
    float* Sr = Ssc + gi * (S_ROWS * S_ROWS) + row * S_ROWS;
    float mx = -3.0e38f;
    for (int j = 0; j < S_ROWS; ++j) mx = fmaxf(mx, Sr[j]);
    float s = 0.0f;
    for (int j = 0; j < S_ROWS; ++j) { float e = __expf(Sr[j] - mx); Sr[j] = e; s += e; }
    const float inv = __builtin_amdgcn_rcpf(s);
    for (int j = 0; j < S_ROWS; ++j) Sr[j] *= inv;
  }
  __syncthreads();
  if (tid < GPB * S_ROWS) {
    const int gi = tid >> 5, j = tid & 31;
    float s = 0.0f;
    for (int i = 0; i < S_ROWS; ++i) s += Ssc[gi * (S_ROWS * S_ROWS) + i * S_ROWS + j];
    wcol[tid] = s;
  }
  __syncthreads();

  // ---- 8) aggregate: agg[d] = sum_j w[j] * x[j][d]   (xs = S1, xv = V1) ----
  for (int idx = tid; idx < GPB * SDIM; idx += BLOCK) {
    const int gi = idx >> 7, c = idx & (SDIM - 1);
    float s = 0.0f;
    for (int j = 0; j < S_ROWS; ++j)
      s = fmaf(wcol[gi * S_ROWS + j], (float)S1[(gi * S_ROWS + j) * SDIM + c], s);
    aggS[idx] = s;
  }
  for (int idx = tid; idx < GPB * VDIM; idx += BLOCK) {
    const int gi = idx / VDIM, c = idx % VDIM;
    float s = 0.0f;
    for (int j = 0; j < S_ROWS; ++j)
      s = fmaf(wcol[gi * S_ROWS + j], (float)V1[(gi * S_ROWS + j) * VDIM + c], s);
    aggV[idx] = s;
  }
  __syncthreads();

  // ---- 9) sa rho MLPs ----
  for (int idx = tid; idx < GPB * SDIM; idx += BLOCK) {
    const int gi = idx >> 7, n = idx & (SDIM - 1);
    rtmp[idx] = silu_f(dotN(P.w_sa_rho_s0 + n * SDIM, aggS + gi * SDIM, SDIM) + P.bb_sa_rho_s0[n]);
  }
  __syncthreads();
  for (int idx = tid; idx < GPB * SDIM; idx += BLOCK) {
    const int gi = idx >> 7, n = idx & (SDIM - 1);
    P.out[OUT_SA_S + (size_t)(g0 + gi) * SDIM + n] =
        silu_f(dotN(P.w_sa_rho_s1 + n * SDIM, rtmp + gi * SDIM, SDIM) + P.bb_sa_rho_s1[n]);
  }
  __syncthreads();
  for (int idx = tid; idx < GPB * VDIM; idx += BLOCK) {
    const int gi = idx / VDIM, n = idx % VDIM;
    rtmp[idx] = silu_f(dotN(P.w_sa_rho_v0 + n * VDIM, aggV + gi * VDIM, VDIM) + P.bb_sa_rho_v0[n]);
  }
  __syncthreads();
  for (int idx = tid; idx < GPB * VDIM; idx += BLOCK) {
    const int gi = idx / VDIM, n = idx % VDIM;
    P.out[OUT_SA_V + (size_t)(g0 + gi) * VDIM + n] =
        silu_f(dotN(P.w_sa_rho_v1 + n * VDIM, rtmp + gi * VDIM, VDIM) + P.bb_sa_rho_v1[n]);
  }
}

__global__ void cvt_f32_f16_kernel(const float* __restrict__ src,
                                   _Float16* __restrict__ dst, int n) {
  int i = blockIdx.x * blockDim.x + threadIdx.x;
  if (i < n) dst[i] = (_Float16)src[i];
}

// Workspace layout in _Float16 units
#define WS_DS_PHI_S0 0
#define WS_DS_PHI_S1 16384
#define WS_SA_PHI_S0 32768
#define WS_SA_PHI_S1 49152
#define WS_ATTN_S    65536
#define WS_DS_PHI_V0 81920
#define WS_DS_PHI_V1 229376
#define WS_SA_PHI_V0 376832
#define WS_SA_PHI_V1 524288
#define WS_ATTN_V    671744

extern "C" void kernel_launch(void* const* d_in, const int* in_sizes, int n_in,
                              void* d_out, int out_size, void* d_ws, size_t ws_size,
                              hipStream_t stream) {
  // param leaf order: jax tree flatten (alphabetical dict keys; 'W' < 'b')
  // 0:x 1:batch_indices
  // 2:attn_s.W 3:attn_s.b 4:attn_v.W 5:attn_v.b
  // 6..9: ds_phi_s [W0,b0,W1,b1]  10..13: ds_phi_v
  // 14..17: ds_rho_s               18..21: ds_rho_v
  // 22..25: sa_phi_s               26..29: sa_phi_v
  // 30..33: sa_rho_s               34..37: sa_rho_v
#define FP(i) ((const float*)d_in[(i)])
  _Float16* ws = (_Float16*)d_ws;

  auto cvt = [&](int idx, int wsoff, int n) {
    cvt_f32_f16_kernel<<<(n + 255) / 256, 256, 0, stream>>>(FP(idx), ws + wsoff, n);
  };
  cvt(6,  WS_DS_PHI_S0, 16384);
  cvt(8,  WS_DS_PHI_S1, 16384);
  cvt(22, WS_SA_PHI_S0, 16384);
  cvt(24, WS_SA_PHI_S1, 16384);
  cvt(2,  WS_ATTN_S,    16384);
  cvt(10, WS_DS_PHI_V0, 147456);
  cvt(12, WS_DS_PHI_V1, 147456);
  cvt(26, WS_SA_PHI_V0, 147456);
  cvt(28, WS_SA_PHI_V1, 147456);
  cvt(4,  WS_ATTN_V,    147456);

  CombParams P;
  P.x = FP(0);
  P.w_ds_phi_s0 = ws + WS_DS_PHI_S0;  P.w_ds_phi_s1 = ws + WS_DS_PHI_S1;
  P.w_sa_phi_s0 = ws + WS_SA_PHI_S0;  P.w_sa_phi_s1 = ws + WS_SA_PHI_S1;
  P.w_attn_s    = ws + WS_ATTN_S;
  P.w_ds_phi_v0 = ws + WS_DS_PHI_V0;  P.w_ds_phi_v1 = ws + WS_DS_PHI_V1;
  P.w_sa_phi_v0 = ws + WS_SA_PHI_V0;  P.w_sa_phi_v1 = ws + WS_SA_PHI_V1;
  P.w_attn_v    = ws + WS_ATTN_V;
  P.b_ds_phi_s0 = FP(7);   P.b_ds_phi_s1 = FP(9);
  P.b_sa_phi_s0 = FP(23);  P.b_sa_phi_s1 = FP(25);
  P.b_attn_s    = FP(3);
  P.b_ds_phi_v0 = FP(11);  P.b_ds_phi_v1 = FP(13);
  P.b_sa_phi_v0 = FP(27);  P.b_sa_phi_v1 = FP(29);
  P.b_attn_v    = FP(5);
  P.w_ds_rho_s0 = FP(14); P.bb_ds_rho_s0 = FP(15);
  P.w_ds_rho_s1 = FP(16); P.bb_ds_rho_s1 = FP(17);
  P.w_ds_rho_v0 = FP(18); P.bb_ds_rho_v0 = FP(19);
  P.w_ds_rho_v1 = FP(20); P.bb_ds_rho_v1 = FP(21);
  P.w_sa_rho_s0 = FP(30); P.bb_sa_rho_s0 = FP(31);
  P.w_sa_rho_s1 = FP(32); P.bb_sa_rho_s1 = FP(33);
  P.w_sa_rho_v0 = FP(34); P.bb_sa_rho_v0 = FP(35);
  P.w_sa_rho_v1 = FP(36); P.bb_sa_rho_v1 = FP(37);
  P.out = (float*)d_out;

  conformer_combiner_kernel<<<NBLOCKS, BLOCK, 0, stream>>>(P);
#undef FP
}
